// BidirectionalMamba_15015205666894
// MI455X (gfx1250) — compile-verified, matrix operand order checked by reading, not run on hardware
//
#include <hip/hip_runtime.h>
#include <hip/hip_bf16.h>

typedef unsigned short u16;
typedef unsigned int   u32;
typedef unsigned long long u64;
typedef __attribute__((ext_vector_type(16))) __bf16 v16bf;
typedef __attribute__((ext_vector_type(8)))  float  v8f;

// ---------- constants (fixed problem size) ----------
#define BSZ   2
#define SEQ   1024
#define DM    1024
#define DI    2048      // d_inner
#define DST   16        // d_state
#define DTR   64        // dt_rank
#define NROW  (BSZ*SEQ) // 2048 rows
#define XPN   96        // dt_rank + 2*d_state
#define XPND  128       // padded x-proj N / dbl leading dim (tile-divisible)

// ---------- helpers ----------
__device__ __forceinline__ u16 f2bf(float f) {
    u32 x = __float_as_uint(f);
    x += 0x7FFFu + ((x >> 16) & 1u);   // round-to-nearest-even
    return (u16)(x >> 16);
}
__device__ __forceinline__ float bf2f(u16 v) {
    return __uint_as_float(((u32)v) << 16);
}

// Async global -> LDS DMA (CDNA5): bypasses VGPRs, tracked by ASYNCcnt.
__device__ __forceinline__ void async_copy_b128(u32 lds_off, const u16* gptr) {
    asm volatile("global_load_async_to_lds_b128 %0, %1, off"
                 :
                 : "v"(lds_off), "v"((u64)(uintptr_t)gptr)
                 : "memory");
}

// ======================================================================
// LayerNorm: one block per row (1024 cols), output bf16
// ======================================================================
__global__ __launch_bounds__(256) void layernorm_kernel(
    const float* __restrict__ x, const float* __restrict__ g,
    const float* __restrict__ b, u16* __restrict__ out)
{
    __shared__ float red[256];
    const int row = blockIdx.x;
    const float* xr = x + (size_t)row * DM;
    float s = 0.f;
    for (int i = threadIdx.x; i < DM; i += 256) s += xr[i];
    red[threadIdx.x] = s; __syncthreads();
    for (int o = 128; o > 0; o >>= 1) { if (threadIdx.x < o) red[threadIdx.x] += red[threadIdx.x + o]; __syncthreads(); }
    const float mean = red[0] * (1.f / DM);
    __syncthreads();
    float s2 = 0.f;
    for (int i = threadIdx.x; i < DM; i += 256) { float d = xr[i] - mean; s2 += d * d; }
    red[threadIdx.x] = s2; __syncthreads();
    for (int o = 128; o > 0; o >>= 1) { if (threadIdx.x < o) red[threadIdx.x] += red[threadIdx.x + o]; __syncthreads(); }
    const float rstd = rsqrtf(red[0] * (1.f / DM) + 1e-5f);
    for (int i = threadIdx.x; i < DM; i += 256)
        out[(size_t)row * DM + i] = f2bf((xr[i] - mean) * rstd * g[i] + b[i]);
}

// ======================================================================
// f32 -> bf16 conversion
// ======================================================================
__global__ __launch_bounds__(256) void convert_f32_bf16(
    const float* __restrict__ src, u16* __restrict__ dst, int n)
{
    int i = blockIdx.x * 256 + threadIdx.x;
    if (i < n) dst[i] = f2bf(src[i]);
}

// x-proj weight: convert 96xK -> zero-padded 128xK bf16 (kills all GEMM guards)
__global__ __launch_bounds__(256) void convert_pad_xproj(
    const float* __restrict__ src, u16* __restrict__ dst)
{
    int i = blockIdx.x * 256 + threadIdx.x;   // over XPND*DI
    int row = i >> 11;                        // /DI (2048)
    int col = i & (DI - 1);
    dst[i] = (row < XPN) ? f2bf(src[(size_t)row * DI + col]) : (u16)0;
}

// ======================================================================
// WMMA bf16 GEMM: C[M,N] = sum_k A[m,k] * W[n,k]
//   A: row-major M x K (bf16, lda), W: row-major N x K (bf16, ldb)
//   mode 0: f32 store   mode 1: bf16 store   mode 2: f32 softplus(x+bias[n])
//   Requires: M%128==0, N%128==0, K%32==0 (guaranteed by caller / padding).
//   Double-buffered LDS filled by async LDS-DMA; WMMAs overlap next-slice copy.
// ======================================================================
#define TM 128
#define TN 128
#define TK 32
#define LSTR 48   // LDS row stride in bf16 elems (96 B: 16B-aligned, conflict-spread)

union FragB16 { v16bf v; uint4 q[2]; };

__global__ __launch_bounds__(256) void gemm_bf16_kernel(
    const u16* __restrict__ A, int lda,
    const u16* __restrict__ Bw, int ldb,
    float* __restrict__ Cf, u16* __restrict__ Cb, int ldc,
    int K, const float* __restrict__ bias, int mode)
{
    __shared__ __align__(16) u16 sA[2][TM * LSTR];
    __shared__ __align__(16) u16 sB[2][TN * LSTR];

    const int tid  = threadIdx.x;
    const int lane = tid & 31;
    const int wid  = tid >> 5;
    const int wm   = wid & 3;   // 4 waves along M (32 rows each)
    const int wn   = wid >> 2;  // 2 waves along N (64 cols each)
    const int bm   = blockIdx.y * TM;
    const int bn   = blockIdx.x * TN;

    const int mloc = lane & 15;
    const int koff = (lane >> 4) * 8;   // ISA 16-bit A/B fragment K-offset

    // Cooperative copy slots: 512 chunks of 16B per tile, 2 per thread per tile.
    const int c0 = tid, c1 = tid + 256;
    const int r0 = c0 >> 2, q0 = (c0 & 3) * 8;
    const int r1 = c1 >> 2, q1 = (c1 & 3) * 8;
    const u16* gA0 = A  + (size_t)(bm + r0) * lda + q0;
    const u16* gA1 = A  + (size_t)(bm + r1) * lda + q1;
    const u16* gB0 = Bw + (size_t)(bn + r0) * ldb + q0;
    const u16* gB1 = Bw + (size_t)(bn + r1) * ldb + q1;
    const u32 lA0 = (u32)(uintptr_t)(&sA[0][r0 * LSTR + q0]);
    const u32 lA1 = (u32)(uintptr_t)(&sA[0][r1 * LSTR + q1]);
    const u32 lB0 = (u32)(uintptr_t)(&sB[0][r0 * LSTR + q0]);
    const u32 lB1 = (u32)(uintptr_t)(&sB[0][r1 * LSTR + q1]);
    const u32 bufbytes = (u32)(TM * LSTR * sizeof(u16));   // 12288

    auto issue = [&](int buf, int kk) {
        const u32 boff = buf ? bufbytes : 0u;
        async_copy_b128(lA0 + boff, gA0 + kk);
        async_copy_b128(lA1 + boff, gA1 + kk);
        async_copy_b128(lB0 + boff, gB0 + kk);
        async_copy_b128(lB1 + boff, gB1 + kk);
    };

    v8f acc[2][4] = {};

    issue(0, 0);                      // prologue: fill buffer 0
    const int nk = K / TK;
    for (int ik = 0; ik < nk; ++ik) {
        const int buf = ik & 1;
        __syncthreads();              // all waves done reading buf^1
        if (ik + 1 < nk) {
            issue(buf ^ 1, (ik + 1) * TK);          // prefetch next slice
            asm volatile("s_wait_asynccnt 0x4" ::: "memory");  // this slice landed
        } else {
            asm volatile("s_wait_asynccnt 0x0" ::: "memory");
        }
        __syncthreads();              // every wave's copies for this slice landed

        const u16* pA = sA[buf];
        const u16* pB = sB[buf];
        FragB16 fa[2], fb[4];
        #pragma unroll
        for (int i = 0; i < 2; ++i) {
            int r = wm * 32 + i * 16 + mloc;
            fa[i].q[0] = *(const uint4*)(pA + r * LSTR + koff);
            fa[i].q[1] = *(const uint4*)(pA + r * LSTR + 16 + koff);
        }
        #pragma unroll
        for (int j = 0; j < 4; ++j) {
            int r = wn * 64 + j * 16 + mloc;
            fb[j].q[0] = *(const uint4*)(pB + r * LSTR + koff);
            fb[j].q[1] = *(const uint4*)(pB + r * LSTR + 16 + koff);
        }
        #pragma unroll
        for (int i = 0; i < 2; ++i)
            #pragma unroll
            for (int j = 0; j < 4; ++j)
                acc[i][j] = __builtin_amdgcn_wmma_f32_16x16x32_bf16(
                    false, fa[i].v, false, fb[j].v, (short)0, acc[i][j], false, false);
    }

    // C/D layout: vgpr e -> M = base + e + 8*(lane>>4), N = base + (lane&15)
    #pragma unroll
    for (int i = 0; i < 2; ++i) {
        #pragma unroll
        for (int j = 0; j < 4; ++j) {
            const int mbase = bm + wm * 32 + i * 16 + ((lane >> 4) * 8);
            const int n     = bn + wn * 64 + j * 16 + (lane & 15);
            #pragma unroll
            for (int e = 0; e < 8; ++e) {
                const int m = mbase + e;
                float v = acc[i][j][e];
                const size_t idx = (size_t)m * ldc + n;
                if (mode == 2) {
                    float xv = v + bias[n];
                    Cf[idx] = (xv > 20.f) ? xv : logf(1.f + __expf(xv));
                } else if (mode == 1) {
                    Cb[idx] = f2bf(v);
                } else {
                    Cf[idx] = v;
                }
            }
        }
    }
}

// ======================================================================
// Depthwise causal conv (width 4) + SiLU. Reads xi = xz[:, :, 0:DI] at
// PHYSICAL time, writes xc at LOGICAL time (rev maps logical->physical).
// ======================================================================
__global__ __launch_bounds__(256) void conv_silu_kernel(
    const float* __restrict__ xz, const float* __restrict__ cw,
    const float* __restrict__ cb, float* __restrict__ xcf,
    u16* __restrict__ xcb, int rev)
{
    int idx = blockIdx.x * 256 + threadIdx.x;       // B*SEQ*DI threads
    int d = idx & (DI - 1);
    int t = (idx >> 11) & (SEQ - 1);
    int b = idx >> 21;
    float acc = cb[d];
    #pragma unroll
    for (int k = 0; k < 4; ++k) {
        int ts = t - 3 + k;
        if (ts >= 0) {
            int tp = rev ? (SEQ - 1 - ts) : ts;
            acc += xz[((size_t)(b * SEQ + tp)) * (2 * DI) + d] * cw[d * 4 + k];
        }
    }
    float sx = acc / (1.f + __expf(-acc));          // silu
    size_t o = ((size_t)(b * SEQ + t)) * DI + d;
    xcf[o] = sx;
    xcb[o] = f2bf(sx);
}

// ======================================================================
// Selective scan: one thread per channel d holds h[16] in registers.
// B/C (dbl cols 64..95, bf16, ld=128) broadcast via LDS each step.
// Fuses D-skip and *silu(z), emits bf16 u for out-proj GEMM.
// ======================================================================
__global__ __launch_bounds__(256) void scan_kernel(
    const float* __restrict__ delta, const u16* __restrict__ dbl,
    const float* __restrict__ xcf, const float* __restrict__ xz,
    const float* __restrict__ A_log, const float* __restrict__ Dskip,
    u16* __restrict__ u_out, int rev)
{
    __shared__ float sBC[32];
    const int b = blockIdx.x >> 3;
    const int d = (blockIdx.x & 7) * 256 + threadIdx.x;

    float Ad[DST], h[DST];
    #pragma unroll
    for (int n = 0; n < DST; ++n) { Ad[n] = -__expf(A_log[d * DST + n]); h[n] = 0.f; }
    const float Dd = Dskip[d];

    for (int t = 0; t < SEQ; ++t) {
        const size_t r = (size_t)b * SEQ + t;
        __syncthreads();
        if (threadIdx.x < 32) sBC[threadIdx.x] = bf2f(dbl[r * XPND + DTR + threadIdx.x]);
        __syncthreads();
        const float dl = delta[r * DI + d];
        const float x  = xcf[r * DI + d];
        float y = 0.f;
        #pragma unroll
        for (int n = 0; n < DST; ++n) {
            float dA = __expf(dl * Ad[n]);
            h[n] = dA * h[n] + (dl * sBC[n]) * x;
            y += h[n] * sBC[DST + n];
        }
        const int tp = rev ? (SEQ - 1 - t) : t;
        const float z = xz[((size_t)(b * SEQ + tp)) * (2 * DI) + DI + d];
        const float u = (y + x * Dd) * (z / (1.f + __expf(-z)));
        u_out[r * DI + d] = f2bf(u);
    }
}

// ======================================================================
// Final combine: out = 0.5*(y_fwd + flip(y_bwd, time))
// ======================================================================
__global__ __launch_bounds__(256) void combine_kernel(
    const float* __restrict__ of, const float* __restrict__ ob,
    float* __restrict__ out)
{
    int idx = blockIdx.x * 256 + threadIdx.x;   // B*SEQ*DM
    int m = idx & (DM - 1);
    int t = (idx >> 10) & (SEQ - 1);
    int b = idx >> 20;
    float vb = ob[((size_t)(b * SEQ + (SEQ - 1 - t))) * DM + m];
    out[idx] = 0.5f * (of[idx] + vb);
}

// ======================================================================
// Host launcher
// ======================================================================
extern "C" void kernel_launch(void* const* d_in, const int* in_sizes, int n_in,
                              void* d_out, int out_size, void* d_ws, size_t ws_size,
                              hipStream_t stream)
{
    (void)in_sizes; (void)n_in; (void)out_size; (void)ws_size;
    const float* x    = (const float*)d_in[0];
    const float* ln_g = (const float*)d_in[1];
    const float* ln_b = (const float*)d_in[2];

    char* ws = (char*)d_ws;
    size_t off = 0;
    auto take = [&](size_t bytes) -> char* {
        char* p = ws + off;
        off = (off + bytes + 255) & ~(size_t)255;
        return p;
    };

    u16*   xn_bf = (u16*)  take((size_t)NROW * DM * 2);          // 4 MiB
    u16*   w_in  = (u16*)  take((size_t)(2 * DI) * DM * 2);      // 8 MiB
    u16*   w_xp  = (u16*)  take((size_t)XPND * DI * 2);          // padded 128 rows
    u16*   w_dt  = (u16*)  take((size_t)DI * DTR * 2);
    u16*   w_out = (u16*)  take((size_t)DM * DI * 2);            // 4 MiB
    float* xz    = (float*)take((size_t)NROW * (2 * DI) * 4);    // 32 MiB
    float* xcf   = (float*)take((size_t)NROW * DI * 4);          // 16 MiB
    u16*   xcb   = (u16*)  take((size_t)NROW * DI * 2);          // 8 MiB
    u16*   dblb  = (u16*)  take((size_t)NROW * XPND * 2);        // ld = 128
    float* delta = (float*)take((size_t)NROW * DI * 4);          // 16 MiB
    u16*   u_bf  = (u16*)  take((size_t)NROW * DI * 2);          // 8 MiB
    float* outd0 = (float*)take((size_t)NROW * DM * 4);          // 8 MiB
    float* outd1 = (float*)take((size_t)NROW * DM * 4);          // 8 MiB
    float* outd[2] = { outd0, outd1 };

    layernorm_kernel<<<NROW, 256, 0, stream>>>(x, ln_g, ln_b, xn_bf);

    for (int dir = 0; dir < 2; ++dir) {
        const int base = 3 + dir * 9;
        const float* in_w    = (const float*)d_in[base + 0];
        const float* conv_w  = (const float*)d_in[base + 1];
        const float* conv_b  = (const float*)d_in[base + 2];
        const float* xproj_w = (const float*)d_in[base + 3];
        const float* dt_w    = (const float*)d_in[base + 4];
        const float* dt_b    = (const float*)d_in[base + 5];
        const float* A_log   = (const float*)d_in[base + 6];
        const float* D_skip  = (const float*)d_in[base + 7];
        const float* out_w   = (const float*)d_in[base + 8];

        convert_f32_bf16<<<(2 * DI * DM + 255) / 256, 256, 0, stream>>>(in_w, w_in, 2 * DI * DM);
        convert_pad_xproj<<<(XPND * DI) / 256, 256, 0, stream>>>(xproj_w, w_xp);
        convert_f32_bf16<<<(DI * DTR + 255) / 256, 256, 0, stream>>>(dt_w, w_dt, DI * DTR);
        convert_f32_bf16<<<(DM * DI + 255) / 256, 256, 0, stream>>>(out_w, w_out, DM * DI);

        // xz = xn @ in_w.T : M=2048 N=4096 K=1024
        gemm_bf16_kernel<<<dim3(2 * DI / TN, NROW / TM), 256, 0, stream>>>(
            xn_bf, DM, w_in, DM, xz, nullptr, 2 * DI, DM, nullptr, 0);

        conv_silu_kernel<<<(BSZ * SEQ * DI) / 256, 256, 0, stream>>>(
            xz, conv_w, conv_b, xcf, xcb, dir);

        // dbl = xc @ xproj_w.T : M=2048 N=128(pad) K=2048 (bf16 store, ld=128)
        gemm_bf16_kernel<<<dim3(XPND / TN, NROW / TM), 256, 0, stream>>>(
            xcb, DI, w_xp, DI, nullptr, dblb, XPND, DI, nullptr, 1);

        // delta = softplus(dt @ dt_w.T + dt_b) : M=2048 N=2048 K=64 (A = dbl[:, :64], lda=128)
        gemm_bf16_kernel<<<dim3(DI / TN, NROW / TM), 256, 0, stream>>>(
            dblb, XPND, w_dt, DTR, delta, nullptr, DI, DTR, dt_b, 2);

        scan_kernel<<<BSZ * (DI / 256), 256, 0, stream>>>(
            delta, dblb, xcf, xz, A_log, D_skip, u_bf, dir);

        // out = u @ out_w.T : M=2048 N=1024 K=2048
        gemm_bf16_kernel<<<dim3(DM / TN, NROW / TM), 256, 0, stream>>>(
            u_bf, DI, w_out, DI, outd[dir], nullptr, DM, DI, nullptr, 0);
    }

    combine_kernel<<<(BSZ * SEQ * DM) / 256, 256, 0, stream>>>(
        outd[0], outd[1], (float*)d_out);
}
